// FractalBlockWithWindowAttention_25941602468490
// MI455X (gfx1250) — matrix operand order repaired and checked by reading, not compiled, not run on hardware
//
#include <hip/hip_runtime.h>
#include <hip/hip_bf16.h>

typedef __attribute__((ext_vector_type(16))) _Float16 v16h;
typedef __attribute__((ext_vector_type(8)))  _Float16 v8h;
typedef __attribute__((ext_vector_type(8)))  float    v8f;

typedef __attribute__((ext_vector_type(4))) unsigned int u32x4;
typedef __attribute__((ext_vector_type(8))) int          i32x8;
typedef __attribute__((ext_vector_type(4))) int          i32x4;

#define TOKENS 100352   // 32 * 3136
#define CDIM   256
#define NHEADS 8
#define HD     32
#define WINLEN 49
#define NWIN   2048     // 32 * 8 * 8
#define HIDDEN 512

#if defined(__has_builtin)
#if __has_builtin(__builtin_amdgcn_tensor_load_to_lds) && \
    __has_builtin(__builtin_amdgcn_s_wait_tensorcnt)
#define USE_TDM 1
#endif
#endif
#ifndef USE_TDM
#define USE_TDM 0
#endif

#if USE_TDM
// ---------------------------------------------------------------------------
// Issue one TDM 2D tile load (global -> LDS) per the CDNA5 D# descriptor:
//   tile = 128 rows x 64 f16 elements, global row stride = row_stride
//   LDS rows padded by 4 DWORDs (16B) after every 32 DWORDs (128B row)
// ---------------------------------------------------------------------------
__device__ __forceinline__ void tdm_load_tile_f16(unsigned lds_off,
                                                  const void* gptr,
                                                  int row_stride) {
  unsigned long long ga = (unsigned long long)(size_t)gptr;
  u32x4 g0;
  g0.x = 1u;                                   // count=1 (valid), load
  g0.y = lds_off;                              // D#.lds_addr (bytes)
  g0.z = (unsigned)ga;                         // global_addr[31:0]
  g0.w = (unsigned)(ga >> 32) | (2u << 30);    // global_addr[56:32] | type=2
  i32x8 g1;
  g1[0] = (int)((1u << 16) |                   // data_size = 1 (2 bytes)
                (1u << 20) |                   // pad_enable
                (4u << 22) |                   // pad_interval: 32 DWORDs
                (3u << 25));                   // pad_amount: 4 DWORDs
  g1[1] = (int)(64u << 16);                    // tensor_dim0 = 64
  g1[2] = (int)(128u << 16);                   // tensor_dim1 = 128
  g1[3] = (int)(64u << 16);                    // tile_dim0 = 64
  g1[4] = 128;                                 // tile_dim1 = 128
  g1[5] = row_stride;                          // tensor_dim0_stride
  g1[6] = 0;
  g1[7] = 0;
  i32x4 gz = {0, 0, 0, 0};                     // groups 2/3 unused (2D)
#if defined(__clang_major__) && __clang_major__ >= 23
  i32x8 gz8 = {0, 0, 0, 0, 0, 0, 0, 0};
  __builtin_amdgcn_tensor_load_to_lds(g0, g1, gz, gz, gz8, 0);
#else
  __builtin_amdgcn_tensor_load_to_lds(g0, g1, gz, gz, 0);
#endif
}
#endif

// ---------------------------------------------------------------------------
// fp32 -> f16 weight conversion (one-time pre-pass; weights then live in L2)
// ---------------------------------------------------------------------------
__global__ void cvt_w_kernel(const float* __restrict__ in,
                             _Float16* __restrict__ out, int n4) {
  int i = blockIdx.x * 256 + threadIdx.x;
  if (i < n4) {
    float4 f = *(const float4*)&in[(size_t)i * 4];
    _Float16* o = out + (size_t)i * 4;
    o[0] = (_Float16)f.x; o[1] = (_Float16)f.y;
    o[2] = (_Float16)f.z; o[3] = (_Float16)f.w;
  }
}

// ---------------------------------------------------------------------------
// LayerNorm (wave-per-token). WINDOW=true: destination is window-partition
// order (token t -> source spatial location); WINDOW=false: identity order.
// ---------------------------------------------------------------------------
template <bool WINDOW>
__global__ void ln_kernel(const float* __restrict__ x,
                          const float* __restrict__ g,
                          const float* __restrict__ b,
                          _Float16* __restrict__ out) {
  int wave = threadIdx.x >> 5;
  int lane = threadIdx.x & 31;
  int t = blockIdx.x * 8 + wave;   // destination token index
  if (t >= TOKENS) return;

  size_t src;
  if (WINDOW) {
    int win = t / WINLEN, l = t - win * WINLEN;
    int bb = win >> 6, wi = win & 63;
    int wh = wi >> 3, ww = wi & 7;
    int r = l / 7, cc = l - r * 7;
    int n = (wh * 7 + r) * 56 + (ww * 7 + cc);
    src = (size_t)bb * 3136 + n;
  } else {
    src = (size_t)t;
  }
  const float* row = x + src * CDIM;

  float vals[8];
  float s = 0.f;
#pragma unroll
  for (int i = 0; i < 8; ++i) { vals[i] = row[lane * 8 + i]; s += vals[i]; }
#pragma unroll
  for (int o = 16; o > 0; o >>= 1) s += __shfl_xor(s, o, 32);
  float mean = s * (1.f / 256.f);
  float vs = 0.f;
#pragma unroll
  for (int i = 0; i < 8; ++i) { float d = vals[i] - mean; vs += d * d; }
#pragma unroll
  for (int o = 16; o > 0; o >>= 1) vs += __shfl_xor(vs, o, 32);
  float rstd = rsqrtf(vs * (1.f / 256.f) + 1e-5f);

  _Float16* orow = out + (size_t)t * CDIM;
#pragma unroll
  for (int i = 0; i < 8; ++i) {
    int c = lane * 8 + i;
    orow[c] = (_Float16)((vals[i] - mean) * rstd * g[c] + b[c]);
  }
}

// ---------------------------------------------------------------------------
// Tiled WMMA GEMM:  out[M][N] = A[M][K] (f16) @ W[N][K]^T (f16 weights)
// Block tile 128(M) x 128(N), K-step 64. 256 threads = 8 waves in 4x2 grid,
// each wave owns a 32x64 patch = 2x4 accum tiles, 16 WMMAs per K-step.
// Double-buffered Tensor Data Mover pipeline: DMA of stage i+1 overlaps the
// WMMA compute of stage i (TENSORcnt-tracked, in-order per wave). Falls back
// to synchronous vectorized VMEM staging when TDM builtins are unavailable.
// MODE 0: +bias, scatter to per-(win,head) Q/K/V f16
// MODE 1: +bias, window-merge, +residual(x) -> fp32 x2
// MODE 2: +bias, exact GELU -> f16 hidden
// MODE 3: +bias, +residual(x2) -> fp32 d_out
// ---------------------------------------------------------------------------
template <int MODE>
__global__ void gemm_wmma(const _Float16* __restrict__ A,
                          const _Float16* __restrict__ Bw,
                          const float* __restrict__ bias,
                          int K, int Nout,
                          const float* __restrict__ resid,
                          float* __restrict__ outF,
                          _Float16* __restrict__ outH) {
  __shared__ _Float16 As[2][128][72];  // 64 data halves + 16B pad per row
  __shared__ _Float16 Bs[2][128][72];

  int tid  = threadIdx.x;
  int wave = tid >> 5, lane = tid & 31;
  int m0 = blockIdx.x * 128;
  int n0 = blockIdx.y * 128;
  int wm = (wave >> 1) * 32;     // wave M offset within block tile
  int wn = (wave & 1) * 64;      // wave N offset within block tile

  v8f acc[2][4] = {};

  int frl = lane & 15;
  int kbA = (lane < 16) ? 0 : 8;    // A-fragment half-wave K base
  int kbB = (lane < 16) ? 0 : 8;    // B-fragment half-wave K base

  int nIter = K >> 6;               // K / 64

#if USE_TDM
  // Flat LDS pointer low 32 bits == wave-relative LDS byte offset (ISA 10.2)
  unsigned ldsA[2] = {(unsigned)(size_t)(const void*)&As[0][0][0],
                      (unsigned)(size_t)(const void*)&As[1][0][0]};
  unsigned ldsB[2] = {(unsigned)(size_t)(const void*)&Bs[0][0][0],
                      (unsigned)(size_t)(const void*)&Bs[1][0][0]};
  // Prologue: fill stage 0, start stage 1; in-order completion per wave
  // means tensorcnt<=2 guarantees stage 0 is resident.
  if (wave == 0) {
    tdm_load_tile_f16(ldsA[0], &A[(size_t)m0 * K], K);
    tdm_load_tile_f16(ldsB[0], &Bw[(size_t)n0 * K], K);
    if (nIter > 1) {
      tdm_load_tile_f16(ldsA[1], &A[(size_t)m0 * K + 64], K);
      tdm_load_tile_f16(ldsB[1], &Bw[(size_t)n0 * K + 64], K);
      __builtin_amdgcn_s_wait_tensorcnt(2);
    } else {
      __builtin_amdgcn_s_wait_tensorcnt(0);
    }
  }
  __syncthreads();
#endif

  for (int i = 0; i < nIter; ++i) {
    int cur = i & 1;
#if !USE_TDM
    // Synchronous staging: A and B tiles, each 128x64 f16, b128 vectorized
    int k0 = i * 64;
#pragma unroll
    for (int j = 0; j < 4; ++j) {
      int lin = tid + j * 256;                  // 0..1023
      int row = lin >> 3, cg = (lin & 7) * 8;
      *(uint4*)&As[cur][row][cg] =
          *(const uint4*)&A[(size_t)(m0 + row) * K + k0 + cg];
      *(uint4*)&Bs[cur][row][cg] =
          *(const uint4*)&Bw[(size_t)(n0 + row) * K + k0 + cg];
    }
    __syncthreads();
#endif

#pragma unroll
    for (int ks = 0; ks < 2; ++ks) {
      int kk = ks * 32;
      v16h af[2], bf[4];
#pragma unroll
      for (int ti = 0; ti < 2; ++ti) {
        int row = wm + ti * 16 + frl;
        *(v8h*)&af[ti]       = *(const v8h*)&As[cur][row][kk + kbA];
        *((v8h*)&af[ti] + 1) = *(const v8h*)&As[cur][row][kk + kbA + 16];
      }
#pragma unroll
      for (int tj = 0; tj < 4; ++tj) {
        int col = wn + tj * 16 + frl;
        *(v8h*)&bf[tj]       = *(const v8h*)&Bs[cur][col][kk + kbB];
        *((v8h*)&bf[tj] + 1) = *(const v8h*)&Bs[cur][col][kk + kbB + 16];
      }
#pragma unroll
      for (int ti = 0; ti < 2; ++ti)
#pragma unroll
        for (int tj = 0; tj < 4; ++tj)
          acc[ti][tj] = __builtin_amdgcn_wmma_f32_16x16x32_f16(
              false, af[ti], false, bf[tj], (short)0, acc[ti][tj],
              false, false);
    }
    __syncthreads();                 // all waves done reading buf[cur]

#if USE_TDM
    if (wave == 0) {
      if (i + 2 < nIter) {
        // Refill the buffer just consumed; then wait so that stage i+1
        // (2 older TDM ops) is resident before the release barrier.
        int kn = (i + 2) * 64;
        tdm_load_tile_f16(ldsA[cur], &A[(size_t)m0 * K + kn], K);
        tdm_load_tile_f16(ldsB[cur], &Bw[(size_t)n0 * K + kn], K);
        __builtin_amdgcn_s_wait_tensorcnt(2);
      } else if (i + 1 < nIter) {
        __builtin_amdgcn_s_wait_tensorcnt(0);
      }
    }
    __syncthreads();
#endif
  }

  // Epilogue (element-wise from the documented C/D layout:
  // VGPR v: lanes 0-15 -> M=v, N=lane; lanes 16-31 -> M=v+8, N=lane-16)
  int mo = (lane < 16) ? 0 : 8;
#pragma unroll
  for (int ti = 0; ti < 2; ++ti) {
#pragma unroll
    for (int tj = 0; tj < 4; ++tj) {
#pragma unroll
      for (int vv = 0; vv < 8; ++vv) {
        int gm = m0 + wm + ti * 16 + vv + mo;
        int gn = n0 + wn + tj * 16 + frl;
        float val = acc[ti][tj][vv];
        if (MODE == 0) {
          val += bias[gn];
          int part = gn >> 8, c = gn & 255;
          int head = c >> 5, d = c & 31;
          int win = gm / WINLEN, l = gm - win * WINLEN;
          size_t dst = (size_t)part * ((size_t)TOKENS * CDIM) +
                       (((size_t)win * NHEADS + head) * WINLEN + l) * HD + d;
          outH[dst] = (_Float16)val;
        } else if (MODE == 1) {
          val += bias[gn];
          int win = gm / WINLEN, l = gm - win * WINLEN;
          int bb = win >> 6, wi = win & 63;
          int wh = wi >> 3, ww = wi & 7;
          int r = l / 7, cc = l - r * 7;
          int norig = (wh * 7 + r) * 56 + (ww * 7 + cc);
          size_t oi = ((size_t)bb * 3136 + norig) * CDIM + gn;
          outF[oi] = val + resid[oi];
        } else if (MODE == 2) {
          float t = val + bias[gn];
          float gl = 0.5f * t * (1.f + erff(t * 0.70710678118654752f));
          outH[(size_t)gm * Nout + gn] = (_Float16)gl;
        } else {
          size_t oi = (size_t)gm * Nout + gn;
          outF[oi] = val + bias[gn] + resid[oi];
        }
      }
    }
  }
}

// ---------------------------------------------------------------------------
// Window attention: one block (128 threads = 4 waves) per (window, head).
// Q,K staged row-major; V staged transposed so P@V B-fragments are
// contiguous LDS reads. Scores padded 49 -> 64.
// ---------------------------------------------------------------------------
__global__ void attn_kernel(const _Float16* __restrict__ q,
                            const _Float16* __restrict__ k,
                            const _Float16* __restrict__ v,
                            _Float16* __restrict__ attn_out) {
  __shared__ _Float16 Qs[64][40];
  __shared__ _Float16 Ks[64][40];
  __shared__ _Float16 Vt[32][72];   // Vt[d][l]
  __shared__ float    Ss[64][65];   // stride 65 -> conflict-free row scans
  __shared__ _Float16 Ps[64][72];

  int tid  = threadIdx.x;           // 0..127
  int wave = tid >> 5, lane = tid & 31;
  int wh_id = blockIdx.x;           // win * 8 + head
  size_t base = (size_t)wh_id * WINLEN * HD;

  // Stage Q, K (49x32, zero-padded to 64 rows)
  for (int lin = tid; lin < 64 * 4; lin += 128) {
    int row = lin >> 2, cg = (lin & 3) * 8;
    uint4 zero; zero.x = zero.y = zero.z = zero.w = 0u;
    uint4 qv = zero, kv = zero;
    if (row < WINLEN) {
      qv = *(const uint4*)&q[base + (size_t)row * HD + cg];
      kv = *(const uint4*)&k[base + (size_t)row * HD + cg];
    }
    *(uint4*)&Qs[row][cg] = qv;
    *(uint4*)&Ks[row][cg] = kv;
  }
  // Stage V transposed: Vt[d][l] = V[l][d], zero-padded l in [49,64)
  for (int lin = tid; lin < 32 * 64; lin += 128) {
    int d = lin >> 6, l = lin & 63;
    Vt[d][l] = (l < WINLEN) ? v[base + (size_t)l * HD + d] : (_Float16)0.f;
  }
  __syncthreads();

  int frl = lane & 15;
  int mo  = (lane < 16) ? 0 : 8;
  int ro  = wave * 16;              // this wave's 16-row block
  int kbA = (lane < 16) ? 0 : 8;
  int kbB = (lane < 16) ? 0 : 8;

  // scores = Q(16x32) @ K^T, one WMMA per 16-col tile (K-dim = hd = 32)
  v16h aq;
  {
    int row = ro + frl;
    *(v8h*)&aq       = *(const v8h*)&Qs[row][kbA];
    *((v8h*)&aq + 1) = *(const v8h*)&Qs[row][kbA + 16];
  }
  const float scale = 0.17677669529663687f;  // 1/sqrt(32)
#pragma unroll
  for (int tn = 0; tn < 4; ++tn) {
    v16h bk;
    int col = tn * 16 + frl;
    *(v8h*)&bk       = *(const v8h*)&Ks[col][kbB];
    *((v8h*)&bk + 1) = *(const v8h*)&Ks[col][kbB + 16];
    v8f accS = {};
    accS = __builtin_amdgcn_wmma_f32_16x16x32_f16(
        false, aq, false, bk, (short)0, accS, false, false);
#pragma unroll
    for (int vv = 0; vv < 8; ++vv)
      Ss[ro + vv + mo][tn * 16 + frl] = accS[vv] * scale;
  }
  __syncthreads();

  // Row softmax over valid 49 keys; emit f16 P padded to 64x64
  if (tid < 64) {
    int r = tid;
    if (r < WINLEN) {
      float mx = -1e30f;
      for (int c = 0; c < WINLEN; ++c) mx = fmaxf(mx, Ss[r][c]);
      float sum = 0.f;
      for (int c = 0; c < WINLEN; ++c) {
        float e = __expf(Ss[r][c] - mx);
        sum += e;
        Ss[r][c] = e;
      }
      float inv = 1.f / sum;
      for (int c = 0; c < 64; ++c)
        Ps[r][c] = (c < WINLEN) ? (_Float16)(Ss[r][c] * inv) : (_Float16)0.f;
    } else {
      for (int c = 0; c < 64; ++c) Ps[r][c] = (_Float16)0.f;
    }
  }
  __syncthreads();

  // O(16x32) = P(16x64) @ V(64x32): 2 K-steps x 2 N-tiles
  v8f accO[2] = {};
#pragma unroll
  for (int ks = 0; ks < 2; ++ks) {
    v16h ap;
    int row = ro + frl;
    int kb = ks * 32 + kbA;
    *(v8h*)&ap       = *(const v8h*)&Ps[row][kb];
    *((v8h*)&ap + 1) = *(const v8h*)&Ps[row][kb + 16];
#pragma unroll
    for (int tj = 0; tj < 2; ++tj) {
      v16h bv;
      int col = tj * 16 + frl;      // output channel d
      int kb2 = ks * 32 + kbB;
      *(v8h*)&bv       = *(const v8h*)&Vt[col][kb2];
      *((v8h*)&bv + 1) = *(const v8h*)&Vt[col][kb2 + 16];
      accO[tj] = __builtin_amdgcn_wmma_f32_16x16x32_f16(
          false, ap, false, bv, (short)0, accO[tj], false, false);
    }
  }

  // Write back to window-token-major [win*49+l][256] layout
  int win = wh_id >> 3, head = wh_id & 7;
#pragma unroll
  for (int tj = 0; tj < 2; ++tj) {
#pragma unroll
    for (int vv = 0; vv < 8; ++vv) {
      int m = ro + vv + mo;
      if (m < WINLEN) {
        int d = tj * 16 + frl;
        attn_out[((size_t)win * WINLEN + m) * CDIM + head * HD + d] =
            (_Float16)accO[tj][vv];
      }
    }
  }
}

// ---------------------------------------------------------------------------
extern "C" void kernel_launch(void* const* d_in, const int* in_sizes, int n_in,
                              void* d_out, int out_size, void* d_ws, size_t ws_size,
                              hipStream_t stream) {
  const float* x    = (const float*)d_in[0];
  const float* n1g  = (const float*)d_in[3];
  const float* n1b  = (const float*)d_in[4];
  const float* wqkv = (const float*)d_in[5];
  const float* bqkv = (const float*)d_in[6];
  const float* wout = (const float*)d_in[7];
  const float* bout = (const float*)d_in[8];
  const float* n2g  = (const float*)d_in[9];
  const float* n2b  = (const float*)d_in[10];
  const float* w1   = (const float*)d_in[11];
  const float* b1   = (const float*)d_in[12];
  const float* w2   = (const float*)d_in[13];
  const float* b2   = (const float*)d_in[14];

  char* ws = (char*)d_ws;
  const size_t HB = (size_t)TOKENS * CDIM * sizeof(_Float16);  // 51,380,224 B

  _Float16* xw     = (_Float16*)(ws);               // [0, HB)
  _Float16* qkv    = (_Float16*)(ws + HB);          // [HB, 4HB)  q|k|v
  float*    x2     = (float*)(ws + 4 * HB);         // [4HB, 6HB) fp32
  _Float16* attn   = xw;                            // reuse (xw dead after QKV)
  _Float16* h2     = qkv;                           // reuse q region
  _Float16* hidden = (_Float16*)(ws + 2 * HB);      // reuse k+v regions (2HB)

  // f16 weights after x2: qkv(196608) | out(65536) | w1(131072) | w2(131072)
  _Float16* wqkv16 = (_Float16*)(ws + 6 * HB);
  _Float16* wout16 = wqkv16 + 768 * 256;
  _Float16* w1_16  = wout16 + 256 * 256;
  _Float16* w2_16  = w1_16 + 512 * 256;

  const size_t PART = (size_t)TOKENS * CDIM;        // elements per q/k/v part

  // 0) Convert weights to f16 once (stay resident in 192MB L2 afterwards)
  cvt_w_kernel<<<(768 * 256 / 4 + 255) / 256, 256, 0, stream>>>(wqkv, wqkv16, 768 * 256 / 4);
  cvt_w_kernel<<<(256 * 256 / 4 + 255) / 256, 256, 0, stream>>>(wout, wout16, 256 * 256 / 4);
  cvt_w_kernel<<<(512 * 256 / 4 + 255) / 256, 256, 0, stream>>>(w1, w1_16, 512 * 256 / 4);
  cvt_w_kernel<<<(256 * 512 / 4 + 255) / 256, 256, 0, stream>>>(w2, w2_16, 256 * 512 / 4);

  // 1) LN1 + window partition -> f16
  ln_kernel<true><<<TOKENS / 8, 256, 0, stream>>>(x, n1g, n1b, xw);

  // 2) QKV projection: (100352 x 256) @ (256 x 768)
  gemm_wmma<0><<<dim3(TOKENS / 128, 768 / 128), 256, 0, stream>>>(
      xw, wqkv16, bqkv, CDIM, 768, nullptr, nullptr, qkv);

  // 3) Window attention: 2048 windows x 8 heads
  attn_kernel<<<NWIN * NHEADS, 128, 0, stream>>>(
      qkv, qkv + PART, qkv + 2 * PART, attn);

  // 4) Out projection + window merge + residual -> x2 (fp32)
  gemm_wmma<1><<<dim3(TOKENS / 128, CDIM / 128), 256, 0, stream>>>(
      attn, wout16, bout, CDIM, CDIM, x, x2, nullptr);

  // 5) LN2 -> f16
  ln_kernel<false><<<TOKENS / 8, 256, 0, stream>>>(x2, n2g, n2b, h2);

  // 6) MLP up + exact GELU -> f16 hidden
  gemm_wmma<2><<<dim3(TOKENS / 128, HIDDEN / 128), 256, 0, stream>>>(
      h2, w1_16, b1, CDIM, HIDDEN, nullptr, nullptr, hidden);

  // 7) MLP down + residual -> d_out (fp32)
  gemm_wmma<3><<<dim3(TOKENS / 128, CDIM / 128), 256, 0, stream>>>(
      hidden, w2_16, b2, HIDDEN, CDIM, x2, (float*)d_out, nullptr);
}
